// GLATransformerBlock_10514079940652
// MI455X (gfx1250) — matrix operand order repaired and checked by reading, not compiled, not run on hardware
//
#include <hip/hip_runtime.h>
#include <hip/hip_bf16.h>
#include <stdint.h>

// ---------------------------------------------------------------------------
// GLA transformer block for MI455X (gfx1250, wave32, WMMA).
// GEMMs: v_wmma_f32_16x16x32_bf16, 256x128 block tile, 64x64 per-wave tile,
// double-buffered GLOBAL_LOAD_ASYNC_TO_LDS_B128 staging (ASYNCcnt pipeline).
// ---------------------------------------------------------------------------

typedef __attribute__((ext_vector_type(16))) __bf16 bf16x16;
typedef __attribute__((ext_vector_type(8)))  float  f32x8;

#define C_DIM   1024
#define HID_DIM 4096
#define T_LEN   2048
#define B_DIM   4
#define N_TOK   (T_LEN * B_DIM)
#define DK      64
#define HEADS   16

#define BM 256
#define BN 128
#define BK 32

// LDS tile row stride in bf16 elems: 32 data + 8 pad = 80 bytes (16B aligned,
// and 20 banks -> 16 consecutive rows hit distinct banks).
#define LDS_STRIDE 40

__device__ __forceinline__ unsigned short f2bf(float f) {
  union { float f; unsigned u; } v; v.f = f;
  unsigned r = v.u + 0x7FFFu + ((v.u >> 16) & 1u);
  return (unsigned short)(r >> 16);
}
__device__ __forceinline__ float bf2f(unsigned short h) {
  union { unsigned u; float f; } v; v.u = ((unsigned)h) << 16; return v.f;
}
__device__ __forceinline__ float silu(float x) {
  return x / (1.0f + __expf(-x));
}

// 16-byte async copy global -> LDS (CDNA5 GLOBAL_LOAD_ASYNC_TO_LDS_B128,
// tracked by ASYNCcnt, bypasses VGPR staging).
__device__ __forceinline__ void cp_async16(unsigned short* lds_dst,
                                           const unsigned short* gsrc) {
#if defined(__HIP_DEVICE_COMPILE__)
  unsigned lds_off = (unsigned)(uintptr_t)lds_dst;  // low 32 bits = LDS offset
  asm volatile("global_load_async_to_lds_b128 %0, %1, off"
               :
               : "v"(lds_off), "v"(gsrc)
               : "memory");
#else
  (void)lds_dst; (void)gsrc;
#endif
}

__device__ __forceinline__ void async_wait_all() {
#if defined(__HIP_DEVICE_COMPILE__)
#if __has_builtin(__builtin_amdgcn_s_wait_asynccnt)
  __builtin_amdgcn_s_wait_asynccnt(0);
#else
  asm volatile("s_wait_asynccnt 0" ::: "memory");
#endif
#endif
}

// ---------------------------------------------------------------------------
// fp32 -> bf16 cast (weights)
// ---------------------------------------------------------------------------
__global__ void cast_f32_bf16(const float* __restrict__ in,
                              unsigned short* __restrict__ out, size_t n) {
  for (size_t i = (size_t)blockIdx.x * blockDim.x + threadIdx.x; i < n;
       i += (size_t)gridDim.x * blockDim.x)
    out[i] = f2bf(in[i]);
}

// ---------------------------------------------------------------------------
// RMSNorm: out_bf16[n, c] = x / (rms + eps) * scale
// ---------------------------------------------------------------------------
__global__ __launch_bounds__(256)
void rmsnorm_kernel(const float* __restrict__ x, const float* __restrict__ scale,
                    unsigned short* __restrict__ out) {
  const int n = blockIdx.x;
  const float* xr = x + (size_t)n * C_DIM;
  float ss = 0.0f;
  for (int c = threadIdx.x; c < C_DIM; c += 256) { float v = xr[c]; ss += v * v; }
  __shared__ float red[256];
  red[threadIdx.x] = ss;
  __syncthreads();
  for (int off = 128; off > 0; off >>= 1) {
    if (threadIdx.x < off) red[threadIdx.x] += red[threadIdx.x + off];
    __syncthreads();
  }
  const float rms = __fsqrt_rn(red[0]) * (1.0f / 32.0f);  // /sqrt(1024)
  const float inv = 1.0f / (rms + 1e-6f);
  unsigned short* orow = out + (size_t)n * C_DIM;
  for (int c = threadIdx.x; c < C_DIM; c += 256)
    orow[c] = f2bf(xr[c] * inv * scale[c]);
}

// ---------------------------------------------------------------------------
// Generic WMMA GEMM: Y[N,O] = epilogue(A[N,K] @ W[O,K]^T)
//   MODE 0: outF = acc                       (q/k/v)
//   MODE 1: outF = silu(acc + bias)          (out-gate projection)
//   MODE 2: outF = res + acc                 (wo / down-proj residual add)
//   MODE 3: outH = bf16(silu(acc))           (FFN gate)
//   MODE 4: outH = bf16(bf2f(mul) * acc)     (FFN up * stored silu(gate))
// Block tile 256x128, BK=32, 256 threads (8 waves, 4Mx2N), each wave 64x64:
// 4 A-frags x 4 B-frags -> 16 wmma / 16 ds_load_b128 per K-step.
// Global->LDS staging is double-buffered async-to-LDS (3 x b128 per thread
// per tile), overlapped with the WMMA compute of the previous tile.
// ---------------------------------------------------------------------------
template <int MODE>
__global__ __launch_bounds__(256, 2)
void gemm_wmma(const unsigned short* __restrict__ A,
               const unsigned short* __restrict__ Bw,
               const float* __restrict__ bias,
               const float* __restrict__ res,
               const unsigned short* __restrict__ mul,
               float* __restrict__ outF,
               unsigned short* __restrict__ outH,
               int K, int O) {
  __shared__ unsigned short ldsA[2][BM * LDS_STRIDE];
  __shared__ unsigned short ldsB[2][BN * LDS_STRIDE];

  const int tid  = threadIdx.x;
  const int wave = tid >> 5;
  const int lane = tid & 31;
  const int waveM = wave >> 1;  // 0..3
  const int waveN = wave & 1;   // 0..1
  const int m0 = blockIdx.y * BM;
  const int n0 = blockIdx.x * BN;

  // async staging assignment: A row tid (2 x 16B), B row tid/2 half tid&1
  const int arowL = tid;
  const int browL = tid >> 1;
  const int bhalf = (tid & 1) << 4;

  f32x8 acc[4][4] = {};

  auto issueTile = [&](int kt, int buf) {
    const unsigned short* ga = A + (size_t)(m0 + arowL) * K + kt;
    cp_async16(&ldsA[buf][arowL * LDS_STRIDE + 0],  ga + 0);
    cp_async16(&ldsA[buf][arowL * LDS_STRIDE + 16], ga + 16);
    const unsigned short* gb = Bw + (size_t)(n0 + browL) * K + kt + bhalf;
    cp_async16(&ldsB[buf][browL * LDS_STRIDE + bhalf], gb);
  };

  const int nt = K / BK;
  issueTile(0, 0);

  const int arowBase = waveM * 64 + (lane & 15);
  const int akoff    = (lane >> 4) << 4;
  const int bcolBase = waveN * 64 + (lane & 15);
  const int bkoff    = (lane >> 4) << 4;

  for (int it = 0; it < nt; ++it) {
    async_wait_all();   // tile `it` landed in LDS (this wave's copies)
    __syncthreads();    // everyone's copies landed; prior compute finished
    if (it + 1 < nt) issueTile((it + 1) * BK, (it + 1) & 1);

    const __bf16* la = (const __bf16*)ldsA[it & 1];
    const __bf16* lb = (const __bf16*)ldsB[it & 1];

    // preload 4 B fragments (32x16 bf16 each, k = 16*(lane/16) + e)
    bf16x16 bfr[4];
#pragma unroll
    for (int bn = 0; bn < 4; ++bn) {
      const int brow = bcolBase + bn * 16;
#pragma unroll
      for (int e = 0; e < 16; ++e)
        bfr[bn][e] = lb[brow * LDS_STRIDE + bkoff + e];
    }

#pragma unroll
    for (int am = 0; am < 4; ++am) {
      const int ar = arowBase + am * 16;
      bf16x16 af;  // A fragment 16x32 bf16, k = 16*(lane/16) + e
#pragma unroll
      for (int e = 0; e < 16; ++e) {
        const int kk = e + akoff;
        af[e] = la[ar * LDS_STRIDE + kk];
      }
#pragma unroll
      for (int bn = 0; bn < 4; ++bn)
        acc[am][bn] = __builtin_amdgcn_wmma_f32_16x16x32_bf16(
            false, af, false, bfr[bn], (short)0, acc[am][bn], false, false);
    }
  }

  // Epilogue. C/D layout: VGPR r -> M = r + 8*(lane>>4); N = lane&15.
#pragma unroll
  for (int am = 0; am < 4; ++am) {
#pragma unroll
    for (int bn = 0; bn < 4; ++bn) {
      const int gcol = n0 + waveN * 64 + bn * 16 + (lane & 15);
#pragma unroll
      for (int r = 0; r < 8; ++r) {
        const int grow = m0 + waveM * 64 + am * 16 + ((lane >> 4) << 3) + r;
        float v = acc[am][bn][r];
        const size_t idx = (size_t)grow * O + gcol;
        if (MODE == 0) {
          outF[idx] = v;
        } else if (MODE == 1) {
          if (bias) v += bias[gcol];
          outF[idx] = silu(v);
        } else if (MODE == 2) {
          outF[idx] = res[idx] + v;
        } else if (MODE == 3) {
          outH[idx] = f2bf(silu(v));
        } else if (MODE == 4) {
          outH[idx] = f2bf(bf2f(mul[idx]) * v);
        }
      }
    }
  }
}

// ---------------------------------------------------------------------------
// Gate MLP: hidden = silu(xn @ wg1^T + bg1)  (16)
//           alpha  = sigmoid((hidden @ wg2^T + bg2) / 16)  (1024)
// One block (128 threads) per token; too skinny for WMMA.
// ---------------------------------------------------------------------------
__global__ __launch_bounds__(128)
void gate_mlp_kernel(const unsigned short* __restrict__ xn,
                     const float* __restrict__ wg1, const float* __restrict__ bg1,
                     const float* __restrict__ wg2, const float* __restrict__ bg2,
                     float* __restrict__ alpha) {
  const int n = blockIdx.x;
  const unsigned short* xr = xn + (size_t)n * C_DIM;
  const int j = threadIdx.x >> 3;  // 0..15 hidden unit
  const int g = threadIdx.x & 7;   // 0..7 partial group
  float s = 0.0f;
  for (int c = g; c < C_DIM; c += 8)
    s += bf2f(xr[c]) * wg1[(size_t)j * C_DIM + c];
  __shared__ float part[16][8];
  __shared__ float hid[16];
  part[j][g] = s;
  __syncthreads();
  if (threadIdx.x < 16) {
    float t = bg1[threadIdx.x];
#pragma unroll
    for (int i = 0; i < 8; ++i) t += part[threadIdx.x][i];
    hid[threadIdx.x] = silu(t);
  }
  __syncthreads();
  float* arow = alpha + (size_t)n * C_DIM;
  for (int c = threadIdx.x; c < C_DIM; c += 128) {
    float t = bg2[c];
#pragma unroll
    for (int i = 0; i < 16; ++i) t += hid[i] * wg2[(size_t)c * 16 + i];
    arow[c] = 1.0f / (1.0f + __expf(-t * (1.0f / 16.0f)));
  }
}

// ---------------------------------------------------------------------------
// GLA recurrent scan + fused per-head LayerNorm + out-gate multiply.
// One block per (b,h); 64 threads (2 wave32); thread tid owns state column
// S[:,tid] (64 f32 in registers). LN reduce: shfl_xor within wave + 2-slot
// LDS combine. Writes g (bf16) rows for the wo GEMM and S_final.
// ---------------------------------------------------------------------------
__global__ __launch_bounds__(64)
void gla_scan_kernel(const float* __restrict__ q, const float* __restrict__ k,
                     const float* __restrict__ v, const float* __restrict__ alpha,
                     const float* __restrict__ outgate,
                     unsigned short* __restrict__ g_out,
                     float* __restrict__ s_final) {
  const int bh = blockIdx.x;
  const int b = bh / HEADS;
  const int h = bh % HEADS;
  const int tid = threadIdx.x;  // column index v
  const int wid = tid >> 5;
  const int lane = tid & 31;
  const int head_off = h * DK;

  float S[DK];
#pragma unroll
  for (int i = 0; i < DK; ++i) S[i] = 0.0f;

  __shared__ float sq[DK], sk[DK], sa[DK];
  __shared__ float wsum[2], wsq[2];

  for (int t = 0; t < T_LEN; ++t) {
    const size_t row = ((size_t)t * B_DIM + b) * C_DIM + head_off;
    __syncthreads();  // protect LDS reuse across iterations
    sq[tid] = q[row + tid];
    sk[tid] = k[row + tid];
    sa[tid] = alpha[row + tid];
    const float vv = v[row + tid];
    __syncthreads();

    float acc = 0.0f;
#pragma unroll
    for (int kk = 0; kk < DK; ++kk) {
      S[kk] = sa[kk] * S[kk] + sk[kk] * vv;
      acc += sq[kk] * S[kk];
    }

    // per-head LayerNorm over the 64 columns
    float s1 = acc, s2 = acc * acc;
#pragma unroll
    for (int m = 16; m >= 1; m >>= 1) {
      s1 += __shfl_xor(s1, m, 32);
      s2 += __shfl_xor(s2, m, 32);
    }
    if (lane == 0) { wsum[wid] = s1; wsq[wid] = s2; }
    __syncthreads();
    const float tot  = wsum[0] + wsum[1];
    const float tot2 = wsq[0] + wsq[1];
    const float mu  = tot * (1.0f / DK);
    const float var = tot2 * (1.0f / DK) - mu * mu;
    const float o   = (acc - mu) * __frsqrt_rn(var + 1e-5f);
    g_out[row + tid] = f2bf(o * outgate[row + tid]);
  }

  // S_final[b, h, k, v]
  float* sf = s_final + (size_t)bh * DK * DK;
#pragma unroll
  for (int kk = 0; kk < DK; ++kk) sf[(size_t)kk * DK + tid] = S[kk];
}

// ---------------------------------------------------------------------------
// Host-side orchestration
// ---------------------------------------------------------------------------
extern "C" void kernel_launch(void* const* d_in, const int* in_sizes, int n_in,
                              void* d_out, int out_size, void* d_ws, size_t ws_size,
                              hipStream_t stream) {
  (void)in_sizes; (void)n_in; (void)out_size; (void)ws_size;
  const float* x      = (const float*)d_in[0];
  const float* scale1 = (const float*)d_in[1];
  const float* scale2 = (const float*)d_in[2];
  const float* wq     = (const float*)d_in[3];
  const float* wk     = (const float*)d_in[4];
  const float* wv     = (const float*)d_in[5];
  const float* wg1    = (const float*)d_in[6];
  const float* bg1    = (const float*)d_in[7];
  const float* wg2    = (const float*)d_in[8];
  const float* bg2    = (const float*)d_in[9];
  const float* wog    = (const float*)d_in[10];
  const float* bog    = (const float*)d_in[11];
  const float* wo     = (const float*)d_in[12];
  const float* w_gate = (const float*)d_in[13];
  const float* w_up   = (const float*)d_in[14];
  const float* w_down = (const float*)d_in[15];

  char* ws = (char*)d_ws;
  size_t off = 0;
  auto alloc = [&](size_t bytes) -> void* {
    void* p = ws + off;
    off = (off + bytes + 255) & ~(size_t)255;
    return p;
  };
  const size_t CC = (size_t)C_DIM * C_DIM;
  const size_t HC = (size_t)HID_DIM * C_DIM;
  const size_t NC = (size_t)N_TOK * C_DIM;
  const size_t NH = (size_t)N_TOK * HID_DIM;

  unsigned short* wq_b    = (unsigned short*)alloc(CC * 2);
  unsigned short* wk_b    = (unsigned short*)alloc(CC * 2);
  unsigned short* wv_b    = (unsigned short*)alloc(CC * 2);
  unsigned short* wog_b   = (unsigned short*)alloc(CC * 2);
  unsigned short* wo_b    = (unsigned short*)alloc(CC * 2);
  unsigned short* wgate_b = (unsigned short*)alloc(HC * 2);
  unsigned short* wup_b   = (unsigned short*)alloc(HC * 2);
  unsigned short* wdown_b = (unsigned short*)alloc(HC * 2);
  unsigned short* xn1     = (unsigned short*)alloc(NC * 2);
  float*          qbuf    = (float*)alloc(NC * 4);
  float*          kbuf    = (float*)alloc(NC * 4);
  float*          vbuf    = (float*)alloc(NC * 4);
  float*          abuf    = (float*)alloc(NC * 4);
  float*          ogbuf   = (float*)alloc(NC * 4);
  unsigned short* gbuf    = (unsigned short*)alloc(NC * 2);
  float*          x1      = (float*)alloc(NC * 4);
  unsigned short* n2      = (unsigned short*)alloc(NC * 2);
  unsigned short* Gh      = (unsigned short*)alloc(NH * 2);
  unsigned short* Hh      = (unsigned short*)alloc(NH * 2);

  // 1) cast weights to bf16
  cast_f32_bf16<<<2048, 256, 0, stream>>>(wq, wq_b, CC);
  cast_f32_bf16<<<2048, 256, 0, stream>>>(wk, wk_b, CC);
  cast_f32_bf16<<<2048, 256, 0, stream>>>(wv, wv_b, CC);
  cast_f32_bf16<<<2048, 256, 0, stream>>>(wog, wog_b, CC);
  cast_f32_bf16<<<2048, 256, 0, stream>>>(wo, wo_b, CC);
  cast_f32_bf16<<<2048, 256, 0, stream>>>(w_gate, wgate_b, HC);
  cast_f32_bf16<<<2048, 256, 0, stream>>>(w_up, wup_b, HC);
  cast_f32_bf16<<<2048, 256, 0, stream>>>(w_down, wdown_b, HC);

  // 2) pre-norm
  rmsnorm_kernel<<<N_TOK, 256, 0, stream>>>(x, scale1, xn1);

  const dim3 gC(C_DIM / BN, N_TOK / BM);    // (8, 32)
  const dim3 gH(HID_DIM / BN, N_TOK / BM);  // (32, 32)

  // 3) q/k/v projections (WMMA)
  gemm_wmma<0><<<gC, 256, 0, stream>>>(xn1, wq_b, nullptr, nullptr, nullptr, qbuf, nullptr, C_DIM, C_DIM);
  gemm_wmma<0><<<gC, 256, 0, stream>>>(xn1, wk_b, nullptr, nullptr, nullptr, kbuf, nullptr, C_DIM, C_DIM);
  gemm_wmma<0><<<gC, 256, 0, stream>>>(xn1, wv_b, nullptr, nullptr, nullptr, vbuf, nullptr, C_DIM, C_DIM);

  // 4) decay gates alpha
  gate_mlp_kernel<<<N_TOK, 128, 0, stream>>>(xn1, wg1, bg1, wg2, bg2, abuf);

  // 5) output gate = silu(xn @ wog^T + bog)
  gemm_wmma<1><<<gC, 256, 0, stream>>>(xn1, wog_b, bog, nullptr, nullptr, ogbuf, nullptr, C_DIM, C_DIM);

  // 6) recurrent scan + LN + gate; S_final -> tail of d_out
  float* s_final = (float*)d_out + NC;
  gla_scan_kernel<<<B_DIM * HEADS, 64, 0, stream>>>(qbuf, kbuf, vbuf, abuf, ogbuf, gbuf, s_final);

  // 7) x1 = x + g @ wo^T
  gemm_wmma<2><<<gC, 256, 0, stream>>>(gbuf, wo_b, nullptr, x, nullptr, x1, nullptr, C_DIM, C_DIM);

  // 8) second norm
  rmsnorm_kernel<<<N_TOK, 256, 0, stream>>>(x1, scale2, n2);

  // 9) FFN: G = silu(n2 @ w_gate^T); H = G * (n2 @ w_up^T)
  gemm_wmma<3><<<gH, 256, 0, stream>>>(n2, wgate_b, nullptr, nullptr, nullptr, nullptr, Gh, C_DIM, HID_DIM);
  gemm_wmma<4><<<gH, 256, 0, stream>>>(n2, wup_b, nullptr, nullptr, Gh, nullptr, Hh, C_DIM, HID_DIM);

  // 10) out = x1 + H @ w_down^T  -> d_out
  gemm_wmma<2><<<gC, 256, 0, stream>>>(Hh, wdown_b, nullptr, x1, nullptr, (float*)d_out, nullptr, HID_DIM, C_DIM);
}
